// HSAUltraLongBlock_64261300683313
// MI455X (gfx1250) — matrix operand order repaired and checked by reading, not compiled, not run on hardware
//
#include <hip/hip_runtime.h>
#include <hip/hip_bf16.h>

// ---------------------------------------------------------------------------
// MI455X (gfx1250) implementation.
//  - bf16 WMMA (v_wmma_f32_16x16x32_bf16) for the three GEMMs
//  - GLOBAL_LOAD_ASYNC_TO_LDS_B128 (ASYNCcnt) tile staging, double-buffered
//  - RMSNorm fused with bf16 quantization + deterministic chunk-mean reduce
//  - hidden_states passthrough via async d2d copy
// ---------------------------------------------------------------------------

typedef __attribute__((ext_vector_type(8)))  __bf16 v8bf;
typedef __attribute__((ext_vector_type(16))) __bf16 v16bf;
typedef __attribute__((ext_vector_type(8)))  float  v8f;

#define DMODEL 2048
#define SCHUNK 128
#define NCHUNK 128          // N*C = 2*64
#define NTOK   16384        // N*T
#define KVDIM  512

// round-to-nearest-even f32 -> bf16 (bit level)
__device__ __forceinline__ unsigned short f2bf(float f) {
    union { float f; unsigned int u; } c; c.f = f;
    unsigned int u = c.u;
    u += 0x7FFFu + ((u >> 16) & 1u);
    return (unsigned short)(u >> 16);
}

// gfx1250 async DMA: cache/memory -> LDS, tracked by ASYNCcnt (ISA 15.18.3)
__device__ __forceinline__ void async_b128(unsigned lds_off, const void* gptr) {
    unsigned long long ga = (unsigned long long)(uintptr_t)gptr;
    asm volatile("global_load_async_to_lds_b128 %0, %1, off"
                 :: "v"(lds_off), "v"(ga) : "memory");
}
__device__ __forceinline__ void wait_async0() {
    asm volatile("s_wait_asynccnt 0x0" ::: "memory");
}

// ---------------------------------------------------------------------------
// Weight conversion: Wk|Wv -> wkv_bf16[1024][2048], Wl -> wl_bf16[512][2048]
// ---------------------------------------------------------------------------
__global__ void convert_weights(const float* __restrict__ Wk,
                                const float* __restrict__ Wv,
                                const float* __restrict__ Wl,
                                unsigned short* __restrict__ wkv,
                                unsigned short* __restrict__ wl) {
    const int WSZ = KVDIM * DMODEL;   // 1048576
    int idx = blockIdx.x * blockDim.x + threadIdx.x;
    if (idx < WSZ) {
        wkv[idx] = f2bf(Wk[idx]);
    } else if (idx < 2 * WSZ) {
        wkv[idx] = f2bf(Wv[idx - WSZ]);
    } else if (idx < 3 * WSZ) {
        wl[idx - 2 * WSZ] = f2bf(Wl[idx - 2 * WSZ]);
    }
}

// ---------------------------------------------------------------------------
// RMSNorm per token + bf16 quantize + deterministic chunk mean.
// One 1024-thread block (32 waves) per chunk of 128 tokens.
// ---------------------------------------------------------------------------
__global__ __launch_bounds__(1024) void rmsnorm_chunk(
        const float* __restrict__ x,          // [16384][2048]
        const float* __restrict__ w,          // [2048]
        unsigned short* __restrict__ enc,     // bf16 [16384][2048]
        unsigned short* __restrict__ cmean) { // bf16 [128][2048]
    __shared__ float s_scale[SCHUNK];
    const int cid  = blockIdx.x;              // 0..127
    const int tid  = threadIdx.x;
    const int wave = tid >> 5;
    const int lane = tid & 31;
    const float* xc = x + (size_t)cid * SCHUNK * DMODEL;

    // Phase A: per-token rsqrt(mean(x^2)+eps); each wave handles 4 tokens.
    #pragma unroll
    for (int j = 0; j < 4; ++j) {
        int t = wave * 4 + j;
        const float* xt = xc + (size_t)t * DMODEL;
        float ss = 0.f;
        #pragma unroll 8
        for (int i = 0; i < DMODEL / 32; ++i) {
            float v = xt[i * 32 + lane];
            ss += v * v;
        }
        #pragma unroll
        for (int off = 16; off > 0; off >>= 1)
            ss += __shfl_xor(ss, off, 32);
        if (lane == 0)
            s_scale[t] = rsqrtf(ss * (1.0f / DMODEL) + 1e-6f);
    }
    __syncthreads();

    // Phase B: each wave owns d-slice [wave*64, wave*64+64); loop all tokens.
    const int d0 = wave * 64;
    const float w0 = w[d0 + lane];
    const float w1 = w[d0 + 32 + lane];
    float a0 = 0.f, a1 = 0.f;
    unsigned short* ec = enc + (size_t)cid * SCHUNK * DMODEL;
    for (int t = 0; t < SCHUNK; ++t) {
        float sc = s_scale[t];
        float e0 = xc[(size_t)t * DMODEL + d0 + lane]      * sc * w0;
        float e1 = xc[(size_t)t * DMODEL + d0 + 32 + lane] * sc * w1;
        ec[(size_t)t * DMODEL + d0 + lane]      = f2bf(e0);
        ec[(size_t)t * DMODEL + d0 + 32 + lane] = f2bf(e1);
        a0 += e0;   // deterministic serial accumulation
        a1 += e1;
    }
    unsigned short* cm = cmean + (size_t)cid * DMODEL;
    cm[d0 + lane]      = f2bf(a0 * (1.0f / SCHUNK));
    cm[d0 + 32 + lane] = f2bf(a1 * (1.0f / SCHUNK));
}

// ---------------------------------------------------------------------------
// bf16 GEMM:  C[M][Ntot] = A[M][2048] * B[Ntot][2048]^T   (both K-contiguous)
// Block tile 128x128, 8 waves (4 in M x 2 in N), each wave: 2x4 WMMA C tiles.
// K-tile = 32, double-buffered LDS staged by GLOBAL_LOAD_ASYNC_TO_LDS_B128.
// Columns >= nsplit are written to Cv (the mem_v region).
// ---------------------------------------------------------------------------
#define KT    32
#define NKT   (DMODEL / KT)   // 64
#define LDA   40              // 32 data + 8 pad elements (80B row stride)

__global__ __launch_bounds__(256) void gemm_bf16(
        const unsigned short* __restrict__ A,   // bf16 [M][2048]
        const unsigned short* __restrict__ B,   // bf16 [Ntot][2048]
        float* __restrict__ Ck,
        float* __restrict__ Cv,
        int nsplit) {
    __shared__ unsigned short As[2][128 * LDA];
    __shared__ unsigned short Bs[2][128 * LDA];

    const int tid  = threadIdx.x;
    const int lane = tid & 31;
    const int wv   = tid >> 5;
    const int wm   = wv & 3;          // wave M position (0..3) -> 32 rows each
    const int wn   = wv >> 2;         // wave N position (0..1) -> 64 cols each
    const int m0   = blockIdx.x * 128;
    const int n0   = blockIdx.y * 128;
    const int half = lane >> 4;       // lane 0-15 vs 16-31
    const int l16  = lane & 15;

    v8f acc[2][4];
    #pragma unroll
    for (int i = 0; i < 2; ++i)
        #pragma unroll
        for (int j = 0; j < 4; ++j)
            #pragma unroll
            for (int e = 0; e < 8; ++e) acc[i][j][e] = 0.0f;

    // Staging map: 512 slots of 16B per tile (128 rows x 4 chunks), 4/thread.
    const int r0 = tid >> 2;          // rows 0..63
    const int c0 = tid & 3;
    const int r1 = r0 + 64;           // rows 64..127
    const unsigned short* ap = A + (size_t)m0 * DMODEL;
    const unsigned short* bp = B + (size_t)n0 * DMODEL;

    // per-thread LDS byte offsets (low 32 bits of flat addr == LDS offset)
    const unsigned la0[2] = {
        (unsigned)(uintptr_t)&As[0][r0 * LDA + c0 * 8],
        (unsigned)(uintptr_t)&As[1][r0 * LDA + c0 * 8] };
    const unsigned la1[2] = {
        (unsigned)(uintptr_t)&As[0][r1 * LDA + c0 * 8],
        (unsigned)(uintptr_t)&As[1][r1 * LDA + c0 * 8] };
    const unsigned lb0[2] = {
        (unsigned)(uintptr_t)&Bs[0][r0 * LDA + c0 * 8],
        (unsigned)(uintptr_t)&Bs[1][r0 * LDA + c0 * 8] };
    const unsigned lb1[2] = {
        (unsigned)(uintptr_t)&Bs[0][r1 * LDA + c0 * 8],
        (unsigned)(uintptr_t)&Bs[1][r1 * LDA + c0 * 8] };

    // stage k-tile `kt` into buffer `nb` (4 async 16B DMAs per thread)
    auto stage = [&](int kt, int nb) {
        const int ko = kt * KT;
        async_b128(la0[nb], ap + (size_t)r0 * DMODEL + ko + c0 * 8);
        async_b128(la1[nb], ap + (size_t)r1 * DMODEL + ko + c0 * 8);
        async_b128(lb0[nb], bp + (size_t)r0 * DMODEL + ko + c0 * 8);
        async_b128(lb1[nb], bp + (size_t)r1 * DMODEL + ko + c0 * 8);
    };

    stage(0, 0);
    wait_async0();
    __syncthreads();

    int buf = 0;
    for (int kt = 0; kt < NKT; ++kt) {
        // kick off DMA for the next tile; overlaps with WMMA below
        if (kt + 1 < NKT) stage(kt + 1, buf ^ 1);
        if (kt + 2 < NKT) {  // L2 prefetch two tiles ahead
            int kp = (kt + 2) * KT;
            __builtin_prefetch(ap + (size_t)r0 * DMODEL + kp + c0 * 8, 0, 0);
            __builtin_prefetch(bp + (size_t)r0 * DMODEL + kp + c0 * 8, 0, 0);
        }

        // --- fragments from LDS (ISA 7.12.2 bf16 layouts) ---
        v16bf afrag[2];
        #pragma unroll
        for (int tm = 0; tm < 2; ++tm) {
            // A 16x32: lane(half,l16): K in [half*8,+8) and [16+half*8,+8)
            const unsigned short* p =
                &As[buf][(wm * 32 + tm * 16 + l16) * LDA + half * 8];
            v8bf lo = *(const v8bf*)p;
            v8bf hi = *(const v8bf*)(p + 16);
            afrag[tm] = __builtin_shufflevector(lo, hi,
                0,1,2,3,4,5,6,7,8,9,10,11,12,13,14,15);
        }
        v16bf bfrag[4];
        #pragma unroll
        for (int tn = 0; tn < 4; ++tn) {
            // B 32x16: lane(half,n=l16): K in [half*8,+8) and [16+half*8,+8)
            const unsigned short* p =
                &Bs[buf][(wn * 64 + tn * 16 + l16) * LDA + half * 8];
            v8bf lo = *(const v8bf*)p;
            v8bf hi = *(const v8bf*)(p + 16);
            bfrag[tn] = __builtin_shufflevector(lo, hi,
                0,1,2,3,4,5,6,7,8,9,10,11,12,13,14,15);
        }
        #pragma unroll
        for (int tm = 0; tm < 2; ++tm)
            #pragma unroll
            for (int tn = 0; tn < 4; ++tn)
                acc[tm][tn] = __builtin_amdgcn_wmma_f32_16x16x32_bf16(
                    false, afrag[tm], false, bfrag[tn],
                    (short)0, acc[tm][tn], false, false);

        // drain our own async DMAs, then block-wide barrier: next buffer ready
        wait_async0();
        __syncthreads();
        buf ^= 1;
    }

    // Write-out. C/D layout: VGPR r -> M = r + half*8, N = l16.
    #pragma unroll
    for (int tm = 0; tm < 2; ++tm) {
        #pragma unroll
        for (int tn = 0; tn < 4; ++tn) {
            int col = n0 + wn * 64 + tn * 16 + l16;
            float* base;
            int c;
            if (col < nsplit) { base = Ck; c = col; }
            else              { base = Cv; c = col - nsplit; }
            #pragma unroll
            for (int r = 0; r < 8; ++r) {
                int row = m0 + wm * 32 + tm * 16 + half * 8 + r;
                base[(size_t)row * KVDIM + c] = acc[tm][tn][r];
            }
        }
    }
}

// ---------------------------------------------------------------------------
extern "C" void kernel_launch(void* const* d_in, const int* in_sizes, int n_in,
                              void* d_out, int out_size, void* d_ws, size_t ws_size,
                              hipStream_t stream) {
    const float* hs = (const float*)d_in[0];   // [2,8192,2048]
    const float* nw = (const float*)d_in[1];   // [2048]
    const float* Wk = (const float*)d_in[2];   // [512,2048]
    const float* Wv = (const float*)d_in[3];   // [512,2048]
    const float* Wl = (const float*)d_in[4];   // [512,2048]

    float* out  = (float*)d_out;
    float* o_hs = out;                                     // 33,554,432 floats
    float* o_k  = o_hs + (size_t)NTOK * DMODEL;            //  8,388,608
    float* o_v  = o_k  + (size_t)NTOK * KVDIM;             //  8,388,608
    float* o_lm = o_v  + (size_t)NTOK * KVDIM;             //     65,536

    char* ws = (char*)d_ws;
    unsigned short* wkv   = (unsigned short*)(ws);                        // 4 MB
    unsigned short* wl    = (unsigned short*)(ws + (size_t)4  * 1048576); // 2 MB
    unsigned short* cmean = (unsigned short*)(ws + (size_t)6  * 1048576); // 0.5 MB
    unsigned short* enc   = (unsigned short*)(ws + (size_t)7  * 1048576); // 64 MB

    // 1) hidden_states passthrough (pure bandwidth, 134 MB)
    hipMemcpyAsync(o_hs, hs, (size_t)NTOK * DMODEL * sizeof(float),
                   hipMemcpyDeviceToDevice, stream);

    // 2) weights -> bf16
    convert_weights<<<dim3((3 * KVDIM * DMODEL) / 256), dim3(256), 0, stream>>>(
        Wk, Wv, Wl, wkv, wl);

    // 3) RMSNorm + bf16 enc + chunk means
    rmsnorm_chunk<<<dim3(NCHUNK), dim3(1024), 0, stream>>>(hs, nw, enc, cmean);

    // 4) K/V GEMM: [16384,2048] x [1024,2048]^T, split columns at 512
    gemm_bf16<<<dim3(NTOK / 128, (2 * KVDIM) / 128), dim3(256), 0, stream>>>(
        enc, wkv, o_k, o_v, KVDIM);

    // 5) landmark GEMM: [128,2048] x [512,2048]^T
    gemm_bf16<<<dim3(1, KVDIM / 128), dim3(256), 0, stream>>>(
        cmean, wl, o_lm, o_lm, 1 << 30);
}
